// TransformerDecoderLayer_12489764897383
// MI455X (gfx1250) — compile-verified, matrix operand order checked by reading, not run on hardware
//
#include <hip/hip_runtime.h>

// ---------------- types ----------------
typedef __bf16 bf16;
typedef __attribute__((ext_vector_type(16))) __bf16 v16bf;
typedef __attribute__((ext_vector_type(8)))  __bf16 v8bf;
typedef __attribute__((ext_vector_type(8)))  float   v8f;

#define S_    1024
#define B_    8
#define D_    1024
#define H_    16
#define F_    4096
#define E_    8
#define NTOK  (S_*B_)
#define LOG2E 1.44269504088896340736f

// ---------------- helpers ----------------
__device__ __forceinline__ unsigned short f2bf_bits(float x) {
  unsigned int u = __float_as_uint(x);
  u += 0x7FFFu + ((u >> 16) & 1u);            // round to nearest even
  return (unsigned short)(u >> 16);
}
__device__ __forceinline__ bf16 f2bf(float x) {
  union { unsigned short u; bf16 b; } cv;
  cv.u = f2bf_bits(x);
  return cv.b;
}

// Async DMA: global -> LDS, 16 bytes, tracked by ASYNCcnt (CDNA5 path).
// LDS operand is the workgroup-relative byte address (= low 32 bits of the
// generic shared pointer per the LDS aperture rules).
__device__ __forceinline__ void async_ld_b128(const bf16* lds_ptr, const bf16* gptr) {
  unsigned l = (unsigned)(size_t)lds_ptr;
  asm volatile("global_load_async_to_lds_b128 %0, %1, off"
               :: "v"(l), "v"(gptr) : "memory");
}
__device__ __forceinline__ void wait_async0() {
  asm volatile("s_wait_asynccnt 0x0" ::: "memory");
}

// Load one 16x32-slice bf16 WMMA fragment for this lane.
// p points at (row_of_this_lane, k_chunk_base = half*8); lane holds
// K = base..base+7 and base+16..base+23 (ISA 7.12.2 16-bit layouts; B symmetric).
__device__ __forceinline__ v16bf load_frag16(const bf16* p) {
  union { v16bf v; v8bf h[2]; } u;
  u.h[0] = *(const v8bf*)(p);
  u.h[1] = *(const v8bf*)(p + 16);
  return u.v;
}

__device__ __forceinline__ v8f wmma_bf16(v16bf a, v16bf b, v8f c) {
  return __builtin_amdgcn_wmma_f32_16x16x32_bf16(false, a, false, b, (short)0, c,
                                                 false, false);
}

// ---------------- fp32 -> bf16 converter ----------------
__global__ __launch_bounds__(256)
void cvt_f32_bf16(const float* __restrict__ in, bf16* __restrict__ out, long long n) {
  long long i = (long long)blockIdx.x * blockDim.x + threadIdx.x;
  long long stride = (long long)gridDim.x * blockDim.x;
  for (; i < n; i += stride) out[i] = f2bf(in[i]);
}

// ---------------- generic WMMA GEMM ----------------
// C[m, j] = sum_k A[m,k] * Bw[j,k] + bias[j]     (all matmuls are X @ W^T)
// A: bf16 (M x K) row-major; Bw: bf16 (N x K) row-major.
// Block tile 128(M) x 256(N), K-step 32, LDS double-buffered with
// GLOBAL_LOAD_ASYNC_TO_LDS_B128.  8 waves = wm(2) x wn(4), each 64x64
// (4 M-frags x 4 N-frags -> 16 WMMAs per k-step).
// Rows restricted to [Mstart, Mend); OOB A rows are clamped (outputs masked).
__global__ __launch_bounds__(256)
void gemm_bf16(const bf16* __restrict__ A, int lda,
               const bf16* __restrict__ Bw, int ldb,
               const float* __restrict__ bias,
               float* __restrict__ Cf, bf16* __restrict__ Cb, int ldc,
               int Mstart, int Mend, int K, int relu) {
  __shared__ alignas(16) bf16 sA[2][128 * 40];  // pad 40 halves = 80B (bank-spread)
  __shared__ alignas(16) bf16 sB[2][256 * 40];

  const int tid  = threadIdx.x;
  const int wave = tid >> 5;
  const int lane = tid & 31;
  const int l16  = lane & 15;
  const int hf   = lane >> 4;
  const int wm   = wave & 1;
  const int wn   = wave >> 1;
  const int m0   = Mstart + blockIdx.x * 128;
  const int n0   = blockIdx.y * 256;

  const v8f vz = {0.f,0.f,0.f,0.f,0.f,0.f,0.f,0.f};
  v8f acc[4][4];
  for (int mi = 0; mi < 4; ++mi)
    for (int ni = 0; ni < 4; ++ni) acc[mi][ni] = vz;

  // cooperative-load mapping: thread -> (row = tid>>1, 32B segment = (tid&1)*16 halves)
  const int crow = tid >> 1;            // 0..127
  const int cseg = (tid & 1) * 16;      // halves
  const int mclamp = Mend - 1;
  const int ga_row = (m0 + crow < Mend) ? (m0 + crow) : mclamp;

  auto issue_tile = [&](int buf, int k0) {
    // A tile: 128 rows x 32 halves
    {
      bf16* l = sA[buf] + crow * 40 + cseg;
      const bf16* g = A + (size_t)ga_row * lda + k0 + cseg;
      async_ld_b128(l, g);
      async_ld_b128(l + 8, g + 8);
    }
    // B tile: 256 rows x 32 halves (two rows per thread)
    for (int i = 0; i < 2; ++i) {
      int row = crow + i * 128;
      bf16* l = sB[buf] + row * 40 + cseg;
      const bf16* g = Bw + (size_t)(n0 + row) * ldb + k0 + cseg;
      async_ld_b128(l, g);
      async_ld_b128(l + 8, g + 8);
    }
  };

  const int nk = K / 32;
  issue_tile(0, 0);
  for (int ik = 0; ik < nk; ++ik) {
    const int cur = ik & 1;
    wait_async0();          // my async writes landed
    __syncthreads();        // everyone's landed; everyone done reading other buf
    if (ik + 1 < nk) issue_tile(cur ^ 1, (ik + 1) * 32);

    v16bf af[4], bfr[4];
    for (int mi = 0; mi < 4; ++mi)
      af[mi] = load_frag16(sA[cur] + (wm * 64 + mi * 16 + l16) * 40 + hf * 8);
    for (int ni = 0; ni < 4; ++ni)
      bfr[ni] = load_frag16(sB[cur] + (wn * 64 + ni * 16 + l16) * 40 + hf * 8);
    for (int mi = 0; mi < 4; ++mi)
      for (int ni = 0; ni < 4; ++ni)
        acc[mi][ni] = wmma_bf16(af[mi], bfr[ni], acc[mi][ni]);
  }

  // Epilogue. C layout: lane l -> col l&15; VGPR v -> row v + 8*(l>=16).
  for (int mi = 0; mi < 4; ++mi) {
    for (int ni = 0; ni < 4; ++ni) {
      int col = n0 + wn * 64 + ni * 16 + l16;
      float bv = bias ? bias[col] : 0.0f;
      for (int v = 0; v < 8; ++v) {
        int row = m0 + wm * 64 + mi * 16 + hf * 8 + v;
        if (row < Mend) {
          float x = acc[mi][ni][v] + bv;
          if (relu) x = fmaxf(x, 0.0f);
          if (Cf) Cf[(size_t)row * ldc + col] = x;
          if (Cb) Cb[(size_t)row * ldc + col] = f2bf(x);
        }
      }
    }
  }
}

// ---------------- flash attention ----------------
// qkv: bf16 (NTOK x 3072); token n = s*B_+b; q at col h*64, k at 1024+h*64,
// v at 2048+h*64.  grid: (S_/128, B_*H_); 8 waves x 16 query rows.
// Online softmax, scale 1/8.  O: bf16 (NTOK x 1024), O[n][h*64+d].
__global__ __launch_bounds__(256)
void attn_kernel(const bf16* __restrict__ qkv, bf16* __restrict__ O) {
  __shared__ alignas(16) bf16 sK[32 * 72];        // [key][d], pad 72
  __shared__ alignas(16) bf16 sVT[64 * 40];       // [d][key], pad 40
  __shared__ alignas(16) bf16 sP[8 * 16 * 40];    // per-wave P tile [row][key]

  const int tid  = threadIdx.x;
  const int wave = tid >> 5;
  const int lane = tid & 31;
  const int l16  = lane & 15;
  const int hf   = lane >> 4;
  const int b    = blockIdx.y >> 4;     // H_ == 16
  const int h    = blockIdx.y & 15;
  const int s0   = blockIdx.x * 128 + wave * 16;
  bf16* sPw = sP + wave * (16 * 40);

  // Q A-fragments (row = l16 -> query s0+l16), K chunks 0..31 / 32..63.
  const bf16* qbase = qkv + ((size_t)(s0 + l16) * B_ + b) * 3072 + h * 64;
  v16bf qf0 = load_frag16(qbase + hf * 8);
  v16bf qf1 = load_frag16(qbase + 32 + hf * 8);

  float mrun[8], lrun[8];
  for (int v = 0; v < 8; ++v) { mrun[v] = -3.0e38f; lrun[v] = 0.0f; }
  const v8f vz = {0.f,0.f,0.f,0.f,0.f,0.f,0.f,0.f};
  v8f o[4];
  for (int ni = 0; ni < 4; ++ni) o[ni] = vz;

  const int key = tid >> 3;   // 0..31
  const int seg = tid & 7;    // 0..7

  for (int kt = 0; kt < S_; kt += 32) {
    // cooperative K tile (async DMA to LDS) + transposed V tile
    const size_t tokoff = ((size_t)(kt + key) * B_ + b) * 3072 + h * 64 + seg * 8;
    async_ld_b128(sK + key * 72 + seg * 8, qkv + tokoff + 1024);
    v8bf vv = *(const v8bf*)(qkv + tokoff + 2048);
    for (int i = 0; i < 8; ++i) sVT[(seg * 8 + i) * 40 + key] = vv[i];
    wait_async0();
    __syncthreads();

    // scores: s1 = keys kt..kt+15, s2 = keys kt+16..kt+31
    v8f s1 = vz, s2 = vz;
    s1 = wmma_bf16(qf0, load_frag16(sK + l16 * 72 + hf * 8), s1);
    s1 = wmma_bf16(qf1, load_frag16(sK + l16 * 72 + 32 + hf * 8), s1);
    s2 = wmma_bf16(qf0, load_frag16(sK + (16 + l16) * 72 + hf * 8), s2);
    s2 = wmma_bf16(qf1, load_frag16(sK + (16 + l16) * 72 + 32 + hf * 8), s2);

    // online softmax: rows live at (vgpr v, lane-half); reduce across 16 lanes
    for (int v = 0; v < 8; ++v) {
      float a = s1[v] * 0.125f;
      float c = s2[v] * 0.125f;
      float mx = fmaxf(a, c);
      mx = fmaxf(mx, __shfl_xor(mx, 1));
      mx = fmaxf(mx, __shfl_xor(mx, 2));
      mx = fmaxf(mx, __shfl_xor(mx, 4));
      mx = fmaxf(mx, __shfl_xor(mx, 8));
      float mn   = fmaxf(mrun[v], mx);
      float corr = exp2f((mrun[v] - mn) * LOG2E);
      float p1   = exp2f((a - mn) * LOG2E);
      float p2   = exp2f((c - mn) * LOG2E);
      float rs   = p1 + p2;
      rs += __shfl_xor(rs, 1);
      rs += __shfl_xor(rs, 2);
      rs += __shfl_xor(rs, 4);
      rs += __shfl_xor(rs, 8);
      lrun[v] = lrun[v] * corr + rs;
      mrun[v] = mn;
      o[0][v] *= corr; o[1][v] *= corr; o[2][v] *= corr; o[3][v] *= corr;
      int prow = hf * 8 + v;                       // C-layout row
      sPw[prow * 40 + l16]      = f2bf(p1);
      sPw[prow * 40 + 16 + l16] = f2bf(p2);
    }

    // P (16x32) as A-fragment; V^T columns as B-fragments; O += P @ V
    v16bf pf = load_frag16(sPw + l16 * 40 + hf * 8);
    for (int ni = 0; ni < 4; ++ni) {
      v16bf vf = load_frag16(sVT + (ni * 16 + l16) * 40 + hf * 8);
      o[ni] = wmma_bf16(pf, vf, o[ni]);
    }
    __syncthreads();
  }

  for (int ni = 0; ni < 4; ++ni) {
    int col = h * 64 + ni * 16 + l16;
    for (int v = 0; v < 8; ++v) {
      int srow = s0 + hf * 8 + v;
      O[((size_t)srow * B_ + b) * 1024 + col] = f2bf(o[ni][v] / lrun[v]);
    }
  }
}

// ---------------- residual + LayerNorm ----------------
__global__ __launch_bounds__(256)
void ln_residual(const float* __restrict__ a, const float* __restrict__ r,
                 const float* __restrict__ gamma, const float* __restrict__ beta,
                 float* __restrict__ outf, bf16* __restrict__ outb) {
  __shared__ float redS[16];
  __shared__ float redQ[16];
  const int tid = threadIdx.x;
  const size_t base = (size_t)blockIdx.x * D_;
  float x[4];
  float s = 0.f, q = 0.f;
  for (int i = 0; i < 4; ++i) {
    int idx = i * 256 + tid;
    float v = a[base + idx] + r[base + idx];
    x[i] = v; s += v; q += v * v;
  }
  for (int m = 1; m < 32; m <<= 1) { s += __shfl_xor(s, m); q += __shfl_xor(q, m); }
  int wave = tid >> 5, lane = tid & 31;
  if (lane == 0) { redS[wave] = s; redQ[wave] = q; }
  __syncthreads();
  if (wave == 0) {
    float ss = (lane < 8) ? redS[lane] : 0.f;
    float qq = (lane < 8) ? redQ[lane] : 0.f;
    for (int m = 1; m < 8; m <<= 1) { ss += __shfl_xor(ss, m); qq += __shfl_xor(qq, m); }
    if (lane == 0) { redS[0] = ss; redQ[0] = qq; }
  }
  __syncthreads();
  float mu   = redS[0] * (1.0f / D_);
  float var  = redQ[0] * (1.0f / D_) - mu * mu;
  float rstd = rsqrtf(var + 1e-5f);
  for (int i = 0; i < 4; ++i) {
    int idx = i * 256 + tid;
    float y = (x[i] - mu) * rstd * gamma[idx] + beta[idx];
    if (outf) outf[base + idx] = y;
    if (outb) outb[base + idx] = f2bf(y);
  }
}

// ---------------- host launcher ----------------
extern "C" void kernel_launch(void* const* d_in, const int* in_sizes, int n_in,
                              void* d_out, int out_size, void* d_ws, size_t ws_size,
                              hipStream_t stream) {
  (void)in_sizes; (void)n_in; (void)out_size; (void)ws_size;
  const float* src  = (const float*)d_in[0];
  const float* inW  = (const float*)d_in[1];
  const float* inB  = (const float*)d_in[2];
  const float* outW = (const float*)d_in[3];
  const float* outB = (const float*)d_in[4];
  const float* g1   = (const float*)d_in[5];
  const float* be1  = (const float*)d_in[6];
  const float* g2   = (const float*)d_in[7];
  const float* be2  = (const float*)d_in[8];
  const float* w1   = (const float*)d_in[9];
  const float* b1   = (const float*)d_in[10];
  const float* w2   = (const float*)d_in[11];
  const float* b2   = (const float*)d_in[12];
  float* out = (float*)d_out;

  char* ws = (char*)d_ws;
  size_t cur = 0;
  auto alloc = [&](size_t bytes) -> char* {
    char* p = ws + cur;
    cur += (bytes + 255) & ~(size_t)255;
    return p;
  };
  bf16*  Xb    = (bf16*) alloc((size_t)NTOK * D_ * 2);
  bf16*  Wqkvb = (bf16*) alloc((size_t)3 * D_ * D_ * 2);
  bf16*  Woutb = (bf16*) alloc((size_t)D_ * D_ * 2);
  bf16*  W1b   = (bf16*) alloc((size_t)E_ * F_ * D_ * 2);
  bf16*  W2b   = (bf16*) alloc((size_t)E_ * D_ * F_ * 2);
  bf16*  QKVb  = (bf16*) alloc((size_t)NTOK * 3 * D_ * 2);
  bf16*  Ob    = (bf16*) alloc((size_t)NTOK * D_ * 2);
  float* Y2f   = (float*)alloc((size_t)NTOK * D_ * 4);
  float* X1f   = (float*)alloc((size_t)NTOK * D_ * 4);
  bf16*  X1b   = (bf16*) alloc((size_t)NTOK * D_ * 2);
  bf16*  Hb    = (bf16*) alloc((size_t)NTOK * F_ * 2);
  float* Y3f   = (float*)alloc((size_t)NTOK * D_ * 4);

  // fp32 -> bf16 operand conversion
  cvt_f32_bf16<<<2048, 256, 0, stream>>>(src,  Xb,    (long long)NTOK * D_);
  cvt_f32_bf16<<<2048, 256, 0, stream>>>(inW,  Wqkvb, (long long)3 * D_ * D_);
  cvt_f32_bf16<<<1024, 256, 0, stream>>>(outW, Woutb, (long long)D_ * D_);
  cvt_f32_bf16<<<4096, 256, 0, stream>>>(w1,   W1b,   (long long)E_ * F_ * D_);
  cvt_f32_bf16<<<4096, 256, 0, stream>>>(w2,   W2b,   (long long)E_ * D_ * F_);

  // QKV projection: (8192 x 1024) @ (3072 x 1024)^T -> bf16
  gemm_bf16<<<dim3(NTOK / 128, (3 * D_) / 256), 256, 0, stream>>>(
      Xb, D_, Wqkvb, D_, inB, nullptr, QKVb, 3 * D_, 0, NTOK, D_, 0);

  // flash attention
  attn_kernel<<<dim3(S_ / 128, B_ * H_), 256, 0, stream>>>(QKVb, Ob);

  // out projection -> f32
  gemm_bf16<<<dim3(NTOK / 128, D_ / 256), 256, 0, stream>>>(
      Ob, D_, Woutb, D_, outB, Y2f, nullptr, D_, 0, NTOK, D_, 0);

  // x1 = LN(src + mha)
  ln_residual<<<NTOK, 256, 0, stream>>>(Y2f, src, g1, be1, X1f, X1b);

  // MoE (contiguous token splits, compile-time constants)
  static const int splits[E_] = {6852, 857, 254, 107, 55, 32, 20, 15};
  int o0 = 0;
  for (int e = 0; e < E_; ++e) {
    int n = splits[e];
    int mt = (n + 127) / 128;
    gemm_bf16<<<dim3(mt, F_ / 256), 256, 0, stream>>>(
        X1b, D_, W1b + (size_t)e * F_ * D_, D_, b1 + (size_t)e * F_,
        nullptr, Hb, F_, o0, o0 + n, D_, 1);
    gemm_bf16<<<dim3(mt, D_ / 256), 256, 0, stream>>>(
        Hb, F_, W2b + (size_t)e * D_ * F_, F_, b2 + (size_t)e * D_,
        Y3f, nullptr, D_, o0, o0 + n, F_, 0);
    o0 += n;
  }

  // out = LN(x1 + moe)
  ln_residual<<<NTOK, 256, 0, stream>>>(Y3f, X1f, g2, be2, out, nullptr);
}
